// MultiHeadAttention_73040213835782
// MI455X (gfx1250) — matrix operand order repaired and checked by reading, not compiled, not run on hardware
//
#include <hip/hip_runtime.h>

// ---------------------------------------------------------------------------
// Problem constants (from reference)
// ---------------------------------------------------------------------------
#define DM    1024          // d_model
#define NH    16            // heads
#define DK    64            // d_k
#define BB    4             // batch
#define LSEQ  2048          // sequence length
#define MROWS (BB * LSEQ)   // 8192 rows for the projection GEMMs
#define NCHUNK (LSEQ / 32)  // mask words per row
#define NCH64 (LSEQ / 64)   // 32 key chunks of 64

typedef __attribute__((ext_vector_type(16))) __bf16 v16bf;
typedef __attribute__((ext_vector_type(8)))  float  v8f;
typedef __attribute__((ext_vector_type(4)))  unsigned int evu32x4;
typedef __attribute__((ext_vector_type(8)))  int          evi32x8;
typedef __attribute__((ext_vector_type(4)))  int          evi32x4;

union Frag { v16bf v; uint4 q[2]; };

#if defined(__has_builtin)
#  if __has_builtin(__builtin_amdgcn_tensor_load_to_lds) && \
      __has_builtin(__builtin_amdgcn_s_wait_tensorcnt)
#    define HAVE_TDM 1
#  else
#    define HAVE_TDM 0
#  endif
#else
#  define HAVE_TDM 0
#endif

__device__ __forceinline__ unsigned short bf16_bits(float f) {
    union { __bf16 b; unsigned short s; } u;
    u.b = (__bf16)f;                       // native RNE convert
    return u.s;
}

#if HAVE_TDM
// Issue a TDM 2D tile load: tile_d1 rows x tile_d0 elements (2-byte data),
// row stride = stride_d0 elements, into LDS at lds_addr_bytes.
__device__ __forceinline__ void tdm_load_2d(unsigned lds_addr_bytes,
                                            const void* gaddr,
                                            unsigned tile_d0, unsigned tile_d1,
                                            unsigned stride_d0) {
    unsigned long long ga = (unsigned long long)gaddr;
    evu32x4 g0;
    g0.x = 1u;                                           // count=1, user desc
    g0.y = lds_addr_bytes;                               // lds_addr
    g0.z = (unsigned)(ga & 0xFFFFFFFFu);                 // global_addr[31:0]
    g0.w = (unsigned)((ga >> 32) & 0x1FFFFFFu) | (2u << 30);  // addr[56:32]|type=2
    evi32x8 g1;
    g1[0] = (int)(1u << 16);                             // wg_mask=0, data_size=1 (2B)
    g1[1] = (int)((tile_d0 & 0xFFFFu) << 16);            // tensor_dim0[15:0]
    g1[2] = (int)(((tile_d0 >> 16) & 0xFFFFu) | ((tile_d1 & 0xFFFFu) << 16));
    g1[3] = (int)(((tile_d1 >> 16) & 0xFFFFu) | ((tile_d0 & 0xFFFFu) << 16)); // tile_dim0
    g1[4] = (int)(tile_d1 & 0xFFFFu);                    // tile_dim1, tile_dim2=0
    g1[5] = (int)stride_d0;                              // tensor_dim0_stride[31:0]
    g1[6] = 0;
    g1[7] = 0;
    evi32x4 gz; gz[0] = 0; gz[1] = 0; gz[2] = 0; gz[3] = 0;
#if __clang_major__ >= 23
    evi32x8 g4 = g1 * 0;                                 // zero
    __builtin_amdgcn_tensor_load_to_lds(g0, g1, gz, gz, g4, 0);
#else
    __builtin_amdgcn_tensor_load_to_lds(g0, g1, gz, gz, 0);
#endif
}
#endif

// ---------------------------------------------------------------------------
// fp32 -> bf16 elementwise convert
// ---------------------------------------------------------------------------
__global__ void cvt_f32_bf16_kernel(const float* __restrict__ in,
                                    unsigned short* __restrict__ out, int n) {
    int i = blockIdx.x * blockDim.x + threadIdx.x;
    int stride = gridDim.x * blockDim.x;
    for (; i < n; i += stride) out[i] = bf16_bits(in[i]);
}

// W [K,N] fp32 -> WT [N,K] bf16
__global__ void transpose_cvt_kernel(const float* __restrict__ W,
                                     unsigned short* __restrict__ WT,
                                     int K, int N) {
    int i = blockIdx.x * blockDim.x + threadIdx.x;
    if (i < K * N) {
        int k = i / N, n = i % N;
        WT[(size_t)n * K + k] = bf16_bits(W[i]);
    }
}

// int32 mask -> bitmask: one uint per (row, 32-key chunk); bit j = (mask != 0)
__global__ void mask_compress_kernel(const int* __restrict__ mask,
                                     unsigned int* __restrict__ bits,
                                     int total_words) {
    int w = blockIdx.x * blockDim.x + threadIdx.x;
    if (w >= total_words) return;
    const int* p = mask + (size_t)w * 32;
    unsigned int word = 0;
#pragma unroll
    for (int j = 0; j < 32; j += 4) {
        int4 m = *(const int4*)(p + j);
        word |= (m.x != 0 ? 1u : 0u) << j;
        word |= (m.y != 0 ? 1u : 0u) << (j + 1);
        word |= (m.z != 0 ? 1u : 0u) << (j + 2);
        word |= (m.w != 0 ? 1u : 0u) << (j + 3);
    }
    bits[w] = word;
}

// ---------------------------------------------------------------------------
// GEMM: out = (A[M,K](bf16) * BT[N,K]^T(bf16) + bias) * oscale (f32 accumulate)
// Block: 128 threads = 4 waves in 2x2 over a 64x128 tile; wave tile = 32x64.
// OUT_MODE: 0 = bf16 row-major, 1 = f32 row-major, 2 = bf16 [B][H][DK][LSEQ]
// ---------------------------------------------------------------------------
template <int OUT_MODE>
__global__ void __launch_bounds__(128)
gemm_bf16_kernel(const unsigned short* __restrict__ A,
                 const unsigned short* __restrict__ BT,
                 const float* __restrict__ bias,
                 void* __restrict__ outp, int M, int N, int K, float oscale) {
    const int tid  = threadIdx.x;
    const int wid  = tid >> 5;
    const int lane = tid & 31;
    const int half = lane >> 4;
    const int ln   = lane & 15;
    const int row0 = blockIdx.y * 64  + (wid >> 1) * 32;
    const int col0 = blockIdx.x * 128 + (wid & 1) * 64;

    v8f acc[2][4];
#pragma unroll
    for (int i = 0; i < 2; i++)
#pragma unroll
        for (int j = 0; j < 4; j++)
#pragma unroll
            for (int r = 0; r < 8; r++) acc[i][j][r] = 0.0f;

    for (int k0 = 0; k0 < K; k0 += 32) {
        Frag a[2], b[4];
#pragma unroll
        for (int i = 0; i < 2; i++) {
            const unsigned short* pa =
                A + (size_t)(row0 + i * 16 + ln) * K + k0 + half * 16;
            a[i].q[0] = *(const uint4*)(pa);
            a[i].q[1] = *(const uint4*)(pa + 8);
        }
#pragma unroll
        for (int j = 0; j < 4; j++) {
            const unsigned short* pb =
                BT + (size_t)(col0 + j * 16 + ln) * K + k0 + half * 16;
            b[j].q[0] = *(const uint4*)(pb);
            b[j].q[1] = *(const uint4*)(pb + 8);
        }
#pragma unroll
        for (int i = 0; i < 2; i++)
#pragma unroll
            for (int j = 0; j < 4; j++)
                acc[i][j] = __builtin_amdgcn_wmma_f32_16x16x32_bf16(
                    false, a[i].v, false, b[j].v, (short)0, acc[i][j],
                    false, false);
    }

#pragma unroll
    for (int i = 0; i < 2; i++)
#pragma unroll
        for (int j = 0; j < 4; j++) {
            int col = col0 + j * 16 + ln;
            float bv = bias[col];
#pragma unroll
            for (int r = 0; r < 8; r++) {
                int row = row0 + i * 16 + r + half * 8;
                float v = (acc[i][j][r] + bv) * oscale;
                if constexpr (OUT_MODE == 1) {
                    ((float*)outp)[(size_t)row * N + col] = v;
                } else if constexpr (OUT_MODE == 0) {
                    ((unsigned short*)outp)[(size_t)row * N + col] = bf16_bits(v);
                } else {
                    size_t idx = (((size_t)(row >> 11) * NH + (col >> 6)) * DK +
                                  (col & (DK - 1))) * LSEQ + (row & (LSEQ - 1));
                    ((unsigned short*)outp)[idx] = bf16_bits(v);
                }
            }
        }
}

// ---------------------------------------------------------------------------
// Flash attention. Grid: (LSEQ/64, B*H). Block: 128 threads = 4 waves.
// 64-key chunks. Q pre-scaled by 1/sqrt(dk). K and V chunks staged in LDS by
// the Tensor Data Mover (double-buffered, one barrier/chunk); rowsum(P) via
// WMMA against a ones-fragment; mask as bitmask words (b64 per row per chunk).
// ---------------------------------------------------------------------------
__global__ void __launch_bounds__(128)
attn_kernel(const unsigned short* __restrict__ Qp,
            const unsigned short* __restrict__ Kp,
            const unsigned short* __restrict__ VpT,
            const unsigned int* __restrict__ mbits,
            unsigned short* __restrict__ outp) {
    __shared__ __align__(16) unsigned short Klds[2][64 * DK];  // K chunk [key][d]
    __shared__ __align__(16) unsigned short Vlds[2][DK * 64];  // V^T chunk [d][key]
    __shared__ __align__(16) unsigned short Pst[4][16][72];    // per-wave P staging

    const int tid  = threadIdx.x;
    const int wid  = tid >> 5;
    const int lane = tid & 31;
    const int half = lane >> 4;
    const int ln   = lane & 15;
    const int bh = blockIdx.y;
    const int b = bh >> 4, h = bh & 15;
    const int q0 = blockIdx.x * 64 + wid * 16;
    const size_t vbase = (size_t)(b * NH + h) * DK * LSEQ;
    const size_t kbase = (size_t)b * LSEQ * DM + h * DK;

    // Q A-fragments (Q already scaled by 1/8 in projection epilogue)
    Frag aq[2];
#pragma unroll
    for (int t = 0; t < 2; t++) {
        const unsigned short* p =
            Qp + ((size_t)b * LSEQ + q0 + ln) * DM + h * DK + t * 32 + half * 16;
        aq[t].q[0] = *(const uint4*)(p);
        aq[t].q[1] = *(const uint4*)(p + 8);
    }

    Frag ones;   // bf16 1.0 replicated: rowsum(P) via WMMA
    ones.q[0] = make_uint4(0x3F803F80u, 0x3F803F80u, 0x3F803F80u, 0x3F803F80u);
    ones.q[1] = ones.q[0];

    v8f o[5];    // o[0..3] = output tiles, o[4] = row sums
    float mrow[8];
#pragma unroll
    for (int c = 0; c < 5; c++)
#pragma unroll
        for (int r = 0; r < 8; r++) o[c][r] = 0.0f;
#pragma unroll
    for (int r = 0; r < 8; r++) mrow[r] = -1e30f;

#if HAVE_TDM
    if (wid == 0) {
        tdm_load_2d((unsigned)(size_t)&Klds[0][0], Kp + kbase, DK, 64, DM);
        tdm_load_2d((unsigned)(size_t)&Vlds[0][0], VpT + vbase, 64, DK, LSEQ);
    }
#endif

    for (int kc = 0; kc < NCH64; kc++) {
        const int key0 = kc * 64;
#if HAVE_TDM
        const int kb = kc & 1;
        if (wid == 0) __builtin_amdgcn_s_wait_tensorcnt(0);
        __syncthreads();   // chunk kc visible; all waves done with buffer kb^1
        if (wid == 0 && kc + 1 < NCH64) {
            tdm_load_2d((unsigned)(size_t)&Klds[kb ^ 1][0],
                        Kp + kbase + (size_t)(key0 + 64) * DM, DK, 64, DM);
            tdm_load_2d((unsigned)(size_t)&Vlds[kb ^ 1][0],
                        VpT + vbase + key0 + 64, 64, DK, LSEQ);
        }
#else
        const int kb = 0;
        __syncthreads();
        {   // cooperative copy: 128 threads x 32 elements each, K and V
            int kk = tid >> 1, dp = (tid & 1) * 32;
            const unsigned short* pk = Kp + kbase + (size_t)(key0 + kk) * DM + dp;
            const unsigned short* pv = VpT + vbase + (size_t)kk * LSEQ + key0 + dp;
#pragma unroll
            for (int e = 0; e < 4; e++) {
                *(uint4*)(&Klds[0][kk * DK + dp + e * 8]) = *(const uint4*)(pk + e * 8);
                *(uint4*)(&Vlds[0][kk * 64 + dp + e * 8]) = *(const uint4*)(pv + e * 8);
            }
        }
        __syncthreads();
#endif

        // warm L2 two chunks ahead of the TDM (global_prefetch_b8)
        if (kc + 2 < NCH64) {
            __builtin_prefetch(Kp + kbase + (size_t)(key0 + 128 + lane * 2) * DM, 0, 3);
            __builtin_prefetch(VpT + vbase + (size_t)(lane * 2) * LSEQ + key0 + 128, 0, 3);
        }

        // mask bitmask words: two adjacent words per row -> one b64 load
        uint2 mw[8];
#pragma unroll
        for (int r = 0; r < 8; r++)
            mw[r] = *(const uint2*)(&mbits[((size_t)b * LSEQ + q0 + r + half * 8) *
                                           NCHUNK + 2 * kc]);

        // ---- S = Q K^T : 16x64 as four 16x16 WMMA tiles, K from LDS ----
        v8f s[4];
#pragma unroll
        for (int t = 0; t < 4; t++)
#pragma unroll
            for (int r = 0; r < 8; r++) s[t][r] = 0.0f;
#pragma unroll
        for (int t = 0; t < 4; t++) {
#pragma unroll
            for (int d = 0; d < 2; d++) {
                Frag bk;
                const unsigned short* p =
                    &Klds[kb][(t * 16 + ln) * DK + d * 32 + half * 16];
                bk.q[0] = *(const uint4*)(p);
                bk.q[1] = *(const uint4*)(p + 8);
                s[t] = __builtin_amdgcn_wmma_f32_16x16x32_bf16(
                    false, aq[d].v, false, bk.v, (short)0, s[t], false, false);
            }
        }

        // ---- online softmax over 64 keys; mask applied post-exp ----
#pragma unroll
        for (int r = 0; r < 8; r++) {
            float mx = fmaxf(fmaxf(s[0][r], s[1][r]), fmaxf(s[2][r], s[3][r]));
#pragma unroll
            for (int off = 8; off >= 1; off >>= 1)
                mx = fmaxf(mx, __shfl_xor(mx, off, 32));
            float mn = fmaxf(mrow[r], mx);
            float sc = __expf(mrow[r] - mn);
            mrow[r] = mn;
#pragma unroll
            for (int c = 0; c < 5; c++) o[c][r] *= sc;
#pragma unroll
            for (int t = 0; t < 4; t++) {
                float p = __expf(s[t][r] - mn);
                unsigned w = (t < 2) ? mw[r].x : mw[r].y;
                p = ((w >> ((t & 1) * 16 + ln)) & 1u) ? p : 0.0f;
                Pst[wid][r + half * 8][t * 16 + ln] = bf16_bits(p);
            }
        }

        // ---- reload P (16x64) as two A-frags (per-wave LDS, in-order) ----
        Frag pa[2];
#pragma unroll
        for (int ks = 0; ks < 2; ks++) {
            pa[ks].q[0] = *(const uint4*)(&Pst[wid][ln][ks * 32 + half * 16]);
            pa[ks].q[1] = *(const uint4*)(&Pst[wid][ln][ks * 32 + 8 + half * 16]);
        }

        // ---- O += P x V ; rowsum += P x ones (V from LDS) ----
#pragma unroll
        for (int ks = 0; ks < 2; ks++) {
#pragma unroll
            for (int c = 0; c < 4; c++) {
                Frag bv;
                const unsigned short* p =
                    &Vlds[kb][(c * 16 + ln) * 64 + ks * 32 + half * 16];
                bv.q[0] = *(const uint4*)(p);
                bv.q[1] = *(const uint4*)(p + 8);
                o[c] = __builtin_amdgcn_wmma_f32_16x16x32_bf16(
                    false, pa[ks].v, false, bv.v, (short)0, o[c], false, false);
            }
            o[4] = __builtin_amdgcn_wmma_f32_16x16x32_bf16(
                false, pa[ks].v, false, ones.v, (short)0, o[4], false, false);
        }
    }

    // ---- normalize + store (bf16, feeds the output projection GEMM) ----
#pragma unroll
    for (int r = 0; r < 8; r++) {
        float inv = 1.0f / o[4][r];
        int qrow = q0 + r + half * 8;
#pragma unroll
        for (int c = 0; c < 4; c++) {
            int dcol = h * DK + c * 16 + ln;
            outp[((size_t)b * LSEQ + qrow) * DM + dcol] = bf16_bits(o[c][r] * inv);
        }
    }
}

// ---------------------------------------------------------------------------
// Host-side orchestration
// ---------------------------------------------------------------------------
extern "C" void kernel_launch(void* const* d_in, const int* in_sizes, int n_in,
                              void* d_out, int out_size, void* d_ws, size_t ws_size,
                              hipStream_t stream) {
    const float* q    = (const float*)d_in[0];
    const float* k    = (const float*)d_in[1];
    const float* v    = (const float*)d_in[2];
    const int*   mask = (const int*)  d_in[3];
    const float* wq   = (const float*)d_in[4];
    const float* bq   = (const float*)d_in[5];
    const float* wk   = (const float*)d_in[6];
    const float* bk   = (const float*)d_in[7];
    const float* wv   = (const float*)d_in[8];
    const float* bv   = (const float*)d_in[9];
    const float* wo   = (const float*)d_in[10];
    const float* bo   = (const float*)d_in[11];
    float* out = (float*)d_out;

    const size_t ACT = (size_t)MROWS * DM;   // 8388608 elements
    const size_t WEL = (size_t)DM * DM;      // 1048576 elements
    const int    MASKW = BB * LSEQ * NCHUNK; // 524288 words

    unsigned short* base = (unsigned short*)d_ws;
    unsigned short* Xq  = base;                 // later reused as attn_out
    unsigned short* Xk  = base + ACT;
    unsigned short* Xv  = base + 2 * ACT;
    unsigned short* WqT = base + 3 * ACT;
    unsigned short* WkT = WqT + WEL;
    unsigned short* WvT = WqT + 2 * WEL;
    unsigned short* WoT = WqT + 3 * WEL;
    unsigned short* Qp  = WqT + 4 * WEL;
    unsigned short* Kp  = Qp + ACT;
    unsigned short* VpT = Kp + ACT;             // [B][H][DK][LSEQ]
    unsigned int*  mbits = (unsigned int*)(VpT + ACT);
    unsigned short* attn_out = Xq;              // Xq dead after projections

    // 1) convert activations to bf16; compress mask to bits
    cvt_f32_bf16_kernel<<<4096, 256, 0, stream>>>(q, Xq, (int)ACT);
    cvt_f32_bf16_kernel<<<4096, 256, 0, stream>>>(k, Xk, (int)ACT);
    cvt_f32_bf16_kernel<<<4096, 256, 0, stream>>>(v, Xv, (int)ACT);
    mask_compress_kernel<<<(MASKW + 255) / 256, 256, 0, stream>>>(mask, mbits, MASKW);

    // 2) convert + transpose weights
    transpose_cvt_kernel<<<4096, 256, 0, stream>>>(wq, WqT, DM, DM);
    transpose_cvt_kernel<<<4096, 256, 0, stream>>>(wk, WkT, DM, DM);
    transpose_cvt_kernel<<<4096, 256, 0, stream>>>(wv, WvT, DM, DM);
    transpose_cvt_kernel<<<4096, 256, 0, stream>>>(wo, WoT, DM, DM);

    // 3) projections: Q scaled by 1/sqrt(dk); V head-transposed
    dim3 ggrid(DM / 128, MROWS / 64);
    gemm_bf16_kernel<0><<<ggrid, 128, 0, stream>>>(Xq, WqT, bq, Qp,  MROWS, DM, DM, 0.125f);
    gemm_bf16_kernel<0><<<ggrid, 128, 0, stream>>>(Xk, WkT, bk, Kp,  MROWS, DM, DM, 1.0f);
    gemm_bf16_kernel<2><<<ggrid, 128, 0, stream>>>(Xv, WvT, bv, VpT, MROWS, DM, DM, 1.0f);

    // 4) flash attention (64-key chunks)
    dim3 agrid(LSEQ / 64, BB * NH);
    attn_kernel<<<agrid, 128, 0, stream>>>(Qp, Kp, VpT, mbits, attn_out);

    // 5) output projection (f32 out)
    gemm_bf16_kernel<1><<<ggrid, 128, 0, stream>>>(attn_out, WoT, bo, out, MROWS, DM, DM, 1.0f);
}
